// MDGAT_51376398795230
// MI455X (gfx1250) — matrix operand order repaired and checked by reading, not compiled, not run on hardware
//
#include <hip/hip_runtime.h>
#include <math.h>

// ---------------------------------------------------------------------------
// CDNA5 WMMA helpers (gfx1250, wave32).
// A-matrix 16x32 f16 layout (ISA 7.12.2): lane L: m = L&15, halves 0..7 hold
//   K = (L>>4)*8 .. +7, halves 8..15 hold K = 16 + (L>>4)*8 .. +7.
// Here make_a and make_b both use contiguous k: lane L: m/n = L&15, halves h=0..15 hold
//   K = (L>>4)*16 + h.
// C/D 16x16 f32: VGPR i, lane L -> m = i + (L>>4)*8, n = L&15.
// ---------------------------------------------------------------------------
typedef __attribute__((ext_vector_type(16))) _Float16 v16h;
typedef __attribute__((ext_vector_type(8)))  float    v8f;

#define WMMA_F16(A, Bm, C) \
  __builtin_amdgcn_wmma_f32_16x16x32_f16(false, (A), false, (Bm), (short)0, (C), false, false)

template <typename F>
__device__ inline v16h make_a(F f) {            // f(m_local, k_local) -> float
  const int lane = threadIdx.x & 31;
  const int m  = lane & 15;
  const int kb = (lane >> 4) * 16;
  v16h a;
#pragma unroll
  for (int hh = 0; hh < 8; ++hh) a[hh]     = (_Float16)f(m, kb + hh);
#pragma unroll
  for (int hh = 0; hh < 8; ++hh) a[hh + 8] = (_Float16)f(m, kb + 8 + hh);
  return a;
}

template <typename F>
__device__ inline v16h make_b(F f) {            // f(k_local, n_local) -> float
  const int lane = threadIdx.x & 31;
  const int n  = lane & 15;
  const int kb = (lane >> 4) * 16;
  v16h b;
#pragma unroll
  for (int hh = 0; hh < 16; ++hh) b[hh] = (_Float16)f(kb + hh, n);
  return b;
}

__device__ inline float wave_max(float v) {
#pragma unroll
  for (int o = 16; o; o >>= 1) v = fmaxf(v, __shfl_xor(v, o, 32));
  return v;
}
__device__ inline float wave_min(float v) {
#pragma unroll
  for (int o = 16; o; o >>= 1) v = fminf(v, __shfl_xor(v, o, 32));
  return v;
}
__device__ inline float wave_sum(float v) {
#pragma unroll
  for (int o = 16; o; o >>= 1) v += __shfl_xor(v, o, 32);
  return v;
}
__device__ inline int wave_isum(int v) {
#pragma unroll
  for (int o = 16; o; o >>= 1) v += __shfl_xor(v, o, 32);
  return v;
}

// ---------------------------------------------------------------------------
// LDS-blocked 1x1-conv GEMM:
//   Y[b,co,n] = epilogue( sum_ci Wa[co,ci]*Xa[b,ci,n] (+ sum Wb*Xb) )
// Block: 256 threads = 8 waves.  Tile: 16 (Co) x 128 (N), K-step 32.
// Per K step the block cooperatively stages the W tile (shared by all 8
// waves) and the X panel into LDS *already in WMMA fragment order* as f16,
// using coalesced vector global loads; each wave then reads its fragments
// as two contiguous 32B LDS loads feeding v_wmma_f32_16x16x32_f16.
// Epilogue: +bias, optional BN (g*inv*val + beta), optional ReLU, residual.
// ---------------------------------------------------------------------------
__global__ void k_gemm(const float* __restrict__ Wa, int lda,
                       const float* __restrict__ Xa, int Ka,
                       const float* __restrict__ Wb, int ldb,
                       const float* __restrict__ Xb, int Kb,
                       const float* __restrict__ bias,
                       const float* __restrict__ bn_g,
                       const float* __restrict__ bn_b,
                       int do_relu,
                       const float* __restrict__ resid,
                       float* __restrict__ Y, int Co, int N)
{
  __shared__ __align__(32) _Float16 Af[32][16];      // A fragments (all waves)
  __shared__ __align__(32) _Float16 Bf[8][32][16];   // B fragments per wave

  const int b    = blockIdx.z;
  const int tid  = threadIdx.x;
  const int wave = tid >> 5;
  const int lane = tid & 31;
  const int nblk = blockIdx.x * 128;                 // block n base
  const int n0   = nblk + wave * 16;                 // this wave's n tile
  const int m0   = blockIdx.y * 16;

  v8f acc = {};
  for (int panel = 0; panel < 2; ++panel) {
    const float* Wp = panel ? Wb : Wa;
    if (!Wp) break;
    const int ld = panel ? ldb : lda;
    const int K  = panel ? Kb : Ka;
    const float* Xp = (panel ? Xb : Xa) + (size_t)b * K * N;

    for (int k0 = 0; k0 < K; k0 += 32) {
      __syncthreads();   // protect LDS from previous iteration's readers

      // ---- stage A tile (16 m x 32 k) -> fragment order ----
      {
        const int m = tid >> 4;                      // 0..15
        const int k = (tid & 15) * 2;                // even k, pair (k, k+1)
        const float* src = &Wp[(size_t)(m0 + m) * ld + k0 + k];
        const float x0 = src[0], x1 = src[1];
        const int la = m + 16 * (k >> 4);            // fragment lane
        const int ha = k & 15;                       // fragment half index
        Af[la][ha]     = (_Float16)x0;               // merged ds_store_b32
        Af[la][ha + 1] = (_Float16)x1;
      }
      // ---- stage X panel (32 k x 128 n) -> fragment order ----
      {
        const int k  = (tid & 15) * 2;               // pair of k rows
        const int nc = (tid >> 4) * 8;               // 8 consecutive n
        const float* r0 = &Xp[(size_t)(k0 + k) * N + nblk + nc];
        const float* r1 = r0 + N;
        if (k0 + 32 < K) __builtin_prefetch(&r0[(size_t)32 * N], 0, 0);
#pragma unroll
        for (int j = 0; j < 8; ++j) {
          const int n  = nc + j;
          const int w  = n >> 4;                     // owning wave
          const int li = (n & 15) + 16 * ((k >> 4) & 1);
          const int hi = k & 15;                     // even
          Bf[w][li][hi]     = (_Float16)r0[j];       // merged ds_store_b32
          Bf[w][li][hi + 1] = (_Float16)r1[j];
        }
      }
      __syncthreads();

      const v16h A  = *(const v16h*)&Af[lane][0];          // 2x ds b128
      const v16h Bm = *(const v16h*)&Bf[wave][lane][0];    // 2x ds b128
      acc = WMMA_F16(A, Bm, acc);
    }
  }

  const float BN_INV = 0.9999950000374997f;          // 1/sqrt(1 + 1e-5)
  float* Yb = Y + (size_t)b * Co * N;
  const float* Rb = resid ? resid + (size_t)b * Co * N : nullptr;
  const int n  = lane & 15;
  const int mb = (lane >> 4) * 8;
#pragma unroll
  for (int i = 0; i < 8; ++i) {
    const int m = m0 + mb + i;
    float v = acc[i];
    if (bias) v += bias[m];
    if (bn_g) v = bn_g[m] * BN_INV * v + bn_b[m];
    if (do_relu) v = fmaxf(v, 0.0f);
    if (Rb) v += Rb[(size_t)m * N + n0 + n];
    Yb[(size_t)m * N + n0 + n] = v;
  }
}

// ---------------------------------------------------------------------------
// Fused attention for one (batch, head, 32-query block).
// Channel layout follows reshape(B, dh, H, N): head h, dim d -> channel d*H+h.
// Phase 1: S[32,1024] = (Q^T K)/sqrt(dh) via WMMA into LDS (Q fragment hoisted
//          out of the key-tile loop).
// Phase 2: per row: max/min, top-k threshold by value bisection (topk>0),
//          masked softmax written back to LDS.
// Phase 3: O = P * V^T via WMMA (P re-read from LDS as f16 A-matrix).
// ---------------------------------------------------------------------------
#define AD  128
#define AH  4
#define ADH 32

__global__ void k_attn(const float* __restrict__ q,
                       const float* __restrict__ kk,
                       const float* __restrict__ vv,
                       float* __restrict__ out, int N, int topk)
{
  extern __shared__ float S[];                  // [32][N]
  const int b  = blockIdx.z;
  const int h  = blockIdx.y;
  const int n0 = blockIdx.x * 32;
  const int wave = threadIdx.x >> 5;
  const int lane = threadIdx.x & 31;
  const size_t base = (size_t)b * AD * N;
  const float* qh = q  + base;
  const float* kh = kk + base;
  const float* vh = vv + base;
  const float scale = 0.17677669529663689f;     // 1/sqrt(32)
  const int n  = lane & 15;
  const int mb = (lane >> 4) * 8;

  // ---- Phase 1: scores (A fragment reused across all key tiles) ----
  const int ntile = N / 16;
#pragma unroll
  for (int qt = 0; qt < 2; ++qt) {
    const v16h A = make_a([&](int ml, int kl) {
      return qh[(size_t)(kl * AH + h) * N + n0 + qt * 16 + ml]; });
    for (int kt = wave; kt < ntile; kt += 8) {
      const v16h Bm = make_b([&](int kl, int nl) {
        return kh[(size_t)(kl * AH + h) * N + kt * 16 + nl]; });
      v8f c = {};
      c = WMMA_F16(A, Bm, c);
#pragma unroll
      for (int i = 0; i < 8; ++i)
        S[(size_t)(qt * 16 + mb + i) * N + kt * 16 + n] = c[i] * scale;
    }
  }
  __syncthreads();

  // ---- Phase 2: per-row (top-k) softmax, one wave per 4 rows ----
  for (int rr = 0; rr < 4; ++rr) {
    float* row = S + (size_t)(wave * 4 + rr) * N;
    float mx = -3.0e38f, mn = 3.0e38f;
    for (int j = lane; j < N; j += 32) {
      const float s = row[j];
      mx = fmaxf(mx, s); mn = fminf(mn, s);
    }
    mx = wave_max(mx); mn = wave_min(mn);

    float thr = -3.0e38f;
    if (topk > 0 && topk < N) {
      float lo = mn, hi = mx + 1.0f;            // count(>=lo)=N>=k, count(>=hi)=0<k
      for (int it = 0; it < 32; ++it) {
        const float mid = 0.5f * (lo + hi);
        int cnt = 0;
        for (int j = lane; j < N; j += 32) cnt += (row[j] >= mid);
        cnt = wave_isum(cnt);
        if (cnt >= topk) lo = mid; else hi = mid;
      }
      thr = lo;                                 // k-th largest (ties measure-zero)
    }

    float sum = 0.0f;
    for (int j = lane; j < N; j += 32) {
      const float s = row[j];
      const float p = (s >= thr) ? __expf(s - mx) : 0.0f;
      row[j] = p;
      sum += p;
    }
    sum = wave_sum(sum);
    const float inv = 1.0f / sum;
    for (int j = lane; j < N; j += 32) row[j] *= inv;
  }
  __syncthreads();

  // ---- Phase 3: O = P V^T (waves 0..3: 2 query tiles x 2 d tiles) ----
  if (wave < 4) {
    const int qt = wave >> 1;
    const int dt = wave & 1;
    v8f acc = {};
    for (int mc = 0; mc < N; mc += 32) {
      const v16h A  = make_a([&](int ml, int kl) {
        return S[(size_t)(qt * 16 + ml) * N + mc + kl]; });
      const v16h Bm = make_b([&](int kl, int nl) {
        return vh[(size_t)((dt * 16 + nl) * AH + h) * N + mc + kl]; });
      acc = WMMA_F16(A, Bm, acc);
    }
#pragma unroll
    for (int i = 0; i < 8; ++i) {
      const int qrow = qt * 16 + mb + i;
      const int d    = dt * 16 + n;
      out[base + (size_t)(d * AH + h) * N + n0 + qrow] = acc[i];
    }
  }
}

// ---------------------------------------------------------------------------
// Final matching scores: Z[b,n,m] = sum_d m0[b,d,n]*m1[b,d,m] / sqrt(128)
// ---------------------------------------------------------------------------
__global__ void k_scores(const float* __restrict__ m0, const float* __restrict__ m1,
                         float* __restrict__ Z, int N, int S1)
{
  const int b    = blockIdx.z;
  const int wave = threadIdx.x >> 5;
  const int lane = threadIdx.x & 31;
  const int ct = blockIdx.x * 8 + wave;
  const int rt = blockIdx.y;
  const float* a0 = m0 + (size_t)b * 128 * N;
  const float* a1 = m1 + (size_t)b * 128 * N;
  v8f acc = {};
  for (int k0 = 0; k0 < 128; k0 += 32) {
    const v16h A  = make_a([&](int ml, int kl) { return a0[(size_t)(k0 + kl) * N + rt * 16 + ml]; });
    const v16h Bm = make_b([&](int kl, int nl) { return a1[(size_t)(k0 + kl) * N + ct * 16 + nl]; });
    acc = WMMA_F16(A, Bm, acc);
  }
  const float s = 0.08838834764831845f;         // 1/sqrt(128)
  float* Zb = Z + (size_t)b * S1 * S1;
  const int n  = lane & 15;
  const int mb = (lane >> 4) * 8;
#pragma unroll
  for (int i = 0; i < 8; ++i)
    Zb[(size_t)(rt * 16 + mb + i) * S1 + ct * 16 + n] = acc[i] * s;
}

__global__ void k_bins(float* __restrict__ Z, const float* __restrict__ alpha,
                       int N, int S1)
{
  const int b = blockIdx.y;
  const int i = blockIdx.x * blockDim.x + threadIdx.x;
  if (i < S1) {
    const float a = alpha[0];
    float* Zb = Z + (size_t)b * S1 * S1;
    Zb[(size_t)i * S1 + N] = a;                 // dustbin column (+ corner)
    Zb[(size_t)N * S1 + i] = a;                 // dustbin row
  }
}

__global__ void k_zero(float* __restrict__ p, int n) {
  const int i = blockIdx.x * blockDim.x + threadIdx.x;
  if (i < n) p[i] = 0.0f;
}

// ---------------------------------------------------------------------------
// Sinkhorn half-steps.  log_mu/log_nu are identical since m == n == 1024.
// ---------------------------------------------------------------------------
__device__ inline float blk_max(float v, float* red) {
  v = wave_max(v);
  const int w = threadIdx.x >> 5;
  if ((threadIdx.x & 31) == 0) red[w] = v;
  __syncthreads();
  float r = red[0];
  const int nw = blockDim.x >> 5;
  for (int t = 1; t < nw; ++t) r = fmaxf(r, red[t]);
  __syncthreads();
  return r;
}
__device__ inline float blk_sum(float v, float* red) {
  v = wave_sum(v);
  const int w = threadIdx.x >> 5;
  if ((threadIdx.x & 31) == 0) red[w] = v;
  __syncthreads();
  float r = 0.0f;
  const int nw = blockDim.x >> 5;
  for (int t = 0; t < nw; ++t) r += red[t];
  __syncthreads();
  return r;
}

__global__ void k_row_lse(const float* __restrict__ Z, const float* __restrict__ v,
                          float* __restrict__ u, int S1, int Nn)
{
  __shared__ float red[8];
  const int b = blockIdx.y, i = blockIdx.x;
  const float* row = Z + ((size_t)b * S1 + i) * S1;
  const float* vb  = v + (size_t)b * S1;
  float mx = -3.0e38f;
  for (int j = threadIdx.x; j < S1; j += blockDim.x) mx = fmaxf(mx, row[j] + vb[j]);
  mx = blk_max(mx, red);
  float s = 0.0f;
  for (int j = threadIdx.x; j < S1; j += blockDim.x) s += __expf(row[j] + vb[j] - mx);
  s = blk_sum(s, red);
  if (threadIdx.x == 0) {
    const float lse  = mx + __logf(s);
    const float norm = -__logf((float)(2 * Nn));
    const float lmu  = (i < Nn) ? norm : (__logf((float)Nn) + norm);
    u[(size_t)b * S1 + i] = lmu - lse;
  }
}

__global__ void k_col_lse(const float* __restrict__ Z, const float* __restrict__ u,
                          float* __restrict__ v, int S1, int Nn)
{
  __shared__ float red[8];
  const int b = blockIdx.y, i = blockIdx.x;  // column index
  const float* Zb = Z + (size_t)b * S1 * S1;
  const float* ub = u + (size_t)b * S1;
  float mx = -3.0e38f;
  for (int j = threadIdx.x; j < S1; j += blockDim.x) mx = fmaxf(mx, Zb[(size_t)j * S1 + i] + ub[j]);
  mx = blk_max(mx, red);
  float s = 0.0f;
  for (int j = threadIdx.x; j < S1; j += blockDim.x) s += __expf(Zb[(size_t)j * S1 + i] + ub[j] - mx);
  s = blk_sum(s, red);
  if (threadIdx.x == 0) {
    const float lse  = mx + __logf(s);
    const float norm = -__logf((float)(2 * Nn));
    const float lnu  = (i < Nn) ? norm : (__logf((float)Nn) + norm);
    v[(size_t)b * S1 + i] = lnu - lse;
  }
}

__global__ void k_out(const float* __restrict__ Z, const float* __restrict__ u,
                      const float* __restrict__ v, float* __restrict__ out,
                      int S1, size_t total)
{
  const size_t idx = (size_t)blockIdx.x * blockDim.x + threadIdx.x;
  if (idx >= total) return;
  const size_t per = (size_t)S1 * S1;
  const int b = (int)(idx / per);
  const size_t rem = idx - (size_t)b * per;
  const int i = (int)(rem / S1);
  const int j = (int)(rem - (size_t)i * S1);
  const float LOG2N = 7.6246189861593985f;      // -norm = log(2048)
  out[idx] = Z[idx] + u[(size_t)b * S1 + i] + v[(size_t)b * S1 + j] + LOG2N;
}

// ---------------------------------------------------------------------------
// Host orchestration
// ---------------------------------------------------------------------------
extern "C" void kernel_launch(void* const* d_in, const int* in_sizes, int n_in,
                              void* d_out, int out_size, void* d_ws, size_t ws_size,
                              hipStream_t stream)
{
  const int B = 4, D = 128, N = 1024, H = 4, L = 6, S1 = 1025;
  (void)in_sizes; (void)n_in; (void)out_size; (void)ws_size;

  const float* desc0 = (const float*)d_in[0];
  const float* desc1 = (const float*)d_in[1];
  const float* Wq = (const float*)d_in[2];  const float* bq = (const float*)d_in[3];
  const float* Wk = (const float*)d_in[4];  const float* bk = (const float*)d_in[5];
  const float* Wv = (const float*)d_in[6];  const float* bv = (const float*)d_in[7];
  const float* Wm = (const float*)d_in[8];  const float* bm = (const float*)d_in[9];
  const float* W1 = (const float*)d_in[10]; const float* b1 = (const float*)d_in[11];
  const float* g1 = (const float*)d_in[12]; const float* be1 = (const float*)d_in[13];
  const float* W2 = (const float*)d_in[14]; const float* b2 = (const float*)d_in[15];
  const float* Wf = (const float*)d_in[16]; const float* bf = (const float*)d_in[17];
  const float* alpha = (const float*)d_in[18];
  const int ITERS = 20;   // reference setup_inputs() fixes iters = 20

  float* ws = (float*)d_ws;
  const size_t SD = (size_t)B * D * N;          // 524288 floats
  float* d0A = ws;            float* d1A = ws + SD;
  float* d0B = ws + 2 * SD;   float* d1B = ws + 3 * SD;
  float* qb  = ws + 4 * SD;   float* kb  = ws + 5 * SD;
  float* vb  = ws + 6 * SD;   float* ab  = ws + 7 * SD;
  float* mb  = ws + 8 * SD;
  float* hb  = ws + 9 * SD;                      // [B,256,N] = 2*SD
  float* Z   = ws + 11 * SD;                     // [B,S1,S1]
  float* ub  = Z + (size_t)B * S1 * S1;          // [B,S1]
  float* vbf = ub + (size_t)B * S1;              // [B,S1]

  hipMemcpyAsync(d0A, desc0, SD * sizeof(float), hipMemcpyDeviceToDevice, stream);
  hipMemcpyAsync(d1A, desc1, SD * sizeof(float), hipMemcpyDeviceToDevice, stream);

  const dim3 blk(256);
  const dim3 gP(N / 128, D / 16, B);             // Co=128 GEMMs
  const dim3 gH(N / 128, 256 / 16, B);           // Co=256 GEMM (MLP hidden)
  const dim3 gA(N / 32, H, B);                   // attention
  const size_t attn_lds = (size_t)32 * N * sizeof(float);   // 128 KB

  float* cur0 = d0A; float* cur1 = d1A;
  float* nxt0 = d0B; float* nxt1 = d1B;
  const int KL[6] = {0, 0, 128, 128, 64, 64};    // top-k per layer (0 = full)

  for (int i = 0; i < L; ++i) {
    const int cross = i & 1;                     // 'self','cross' alternating
    const float* Wq_i = Wq + (size_t)i * D * D;  const float* bq_i = bq + (size_t)i * D;
    const float* Wk_i = Wk + (size_t)i * D * D;  const float* bk_i = bk + (size_t)i * D;
    const float* Wv_i = Wv + (size_t)i * D * D;  const float* bv_i = bv + (size_t)i * D;
    const float* Wm_i = Wm + (size_t)i * D * D;  const float* bm_i = bm + (size_t)i * D;
    const float* W1_i = W1 + (size_t)i * 256 * 256;
    const float* b1_i = b1 + (size_t)i * 256;
    const float* g1_i = g1 + (size_t)i * 256;
    const float* be1_i = be1 + (size_t)i * 256;
    const float* W2_i = W2 + (size_t)i * 128 * 256;
    const float* b2_i = b2 + (size_t)i * 128;

    for (int side = 0; side < 2; ++side) {
      const float* x   = side ? cur1 : cur0;
      const float* src = cross ? (side ? cur0 : cur1) : x;
      float* nxt = side ? nxt1 : nxt0;

      k_gemm<<<gP, blk, 0, stream>>>(Wq_i, D, x,   D, nullptr, 0, nullptr, 0,
                                     bq_i, nullptr, nullptr, 0, nullptr, qb, D, N);
      k_gemm<<<gP, blk, 0, stream>>>(Wk_i, D, src, D, nullptr, 0, nullptr, 0,
                                     bk_i, nullptr, nullptr, 0, nullptr, kb, D, N);
      k_gemm<<<gP, blk, 0, stream>>>(Wv_i, D, src, D, nullptr, 0, nullptr, 0,
                                     bv_i, nullptr, nullptr, 0, nullptr, vb, D, N);
      k_attn<<<gA, blk, attn_lds, stream>>>(qb, kb, vb, ab, N, KL[i]);
      k_gemm<<<gP, blk, 0, stream>>>(Wm_i, D, ab,  D, nullptr, 0, nullptr, 0,
                                     bm_i, nullptr, nullptr, 0, nullptr, mb, D, N);
      // h = relu(bn(W1 * [x; msg] + b1)) : dual-panel GEMM, lda = 256
      k_gemm<<<gH, blk, 0, stream>>>(W1_i, 256, x, 128, W1_i + 128, 256, mb, 128,
                                     b1_i, g1_i, be1_i, 1, nullptr, hb, 256, N);
      // nxt = x + W2 * h + b2
      k_gemm<<<gP, blk, 0, stream>>>(W2_i, 256, hb, 256, nullptr, 0, nullptr, 0,
                                     b2_i, nullptr, nullptr, 0, x, nxt, D, N);
    }
    float* t;
    t = cur0; cur0 = nxt0; nxt0 = t;
    t = cur1; cur1 = nxt1; nxt1 = t;
  }

  // Final projections (reuse qb/kb) + matching scores + dustbins
  k_gemm<<<gP, blk, 0, stream>>>(Wf, D, cur0, D, nullptr, 0, nullptr, 0,
                                 bf, nullptr, nullptr, 0, nullptr, qb, D, N);
  k_gemm<<<gP, blk, 0, stream>>>(Wf, D, cur1, D, nullptr, 0, nullptr, 0,
                                 bf, nullptr, nullptr, 0, nullptr, kb, D, N);
  const dim3 gS(8, N / 16, B);
  k_scores<<<gS, blk, 0, stream>>>(qb, kb, Z, N, S1);
  const dim3 gB((S1 + 255) / 256, B);
  k_bins<<<gB, blk, 0, stream>>>(Z, alpha, N, S1);

  // Sinkhorn: u,v = 0; 20 x (row-lse -> col-lse)
  const int nz = B * S1 * 2;                     // ub and vbf are contiguous
  k_zero<<<(nz + 255) / 256, 256, 0, stream>>>(ub, nz);
  const dim3 gL(S1, B);
  for (int it = 0; it < ITERS; ++it) {
    k_row_lse<<<gL, blk, 0, stream>>>(Z, vbf, ub, S1, N);
    k_col_lse<<<gL, blk, 0, stream>>>(Z, ub, vbf, S1, N);
  }

  const size_t total = (size_t)B * S1 * S1;
  k_out<<<(unsigned)((total + 255) / 256), 256, 0, stream>>>(Z, ub, vbf,
                                                             (float*)d_out, S1, total);
}
